// PureCartesianICTDTransformerLayer_26628797235372
// MI455X (gfx1250) — matrix operand order repaired and checked by reading, not compiled, not run on hardware
//
#include <hip/hip_runtime.h>
#include <hip/hip_bf16.h>

// ---------------------------------------------------------------------------
// MI455X (gfx1250) implementation of the equivariant edge-conv layer.
//
// Math simplification: real CG for (l,0,l) is the identity, so only paths
// p=0 (l=0), p=3 (l=1), p=9 (l=2) contribute and
//   edge_feat = concat_l3 [ gates[:,p] * Y_l3[:,k] * M[:,w,p] ]
//   M[e,w,p]  = sum_{u,v} Ai_src[u]*Ai_dst[v]*tp_w[p][u,v,w]
// The FC MLP (10->64->64->15) and the TP contraction (64->96) are cast as
// f16 WMMA GEMMs (v_wmma_f32_16x16x32_f16) over 16-edge row tiles.
// ---------------------------------------------------------------------------

typedef __attribute__((ext_vector_type(16))) _Float16 v16h;
typedef __attribute__((ext_vector_type(8)))  _Float16 v8h;
typedef __attribute__((ext_vector_type(8)))  float    v8f;
typedef __attribute__((ext_vector_type(4)))  float    v4f;

#define NFEAT       288      // 32*1 + 32*3 + 32*5
#define WAVES_PER_BLOCK 8

__device__ __forceinline__ float silu_f(float x) {
  return x / (1.0f + __expf(-x));
}

__device__ __forceinline__ v8f wmma_f16(v16h a, v16h b, v8f c) {
  return __builtin_amdgcn_wmma_f32_16x16x32_f16(false, a, false, b, (short)0, c,
                                                false, false);
}

__device__ __forceinline__ void lds_fence_wave() {
  // wave-local LDS hand-off (producer lanes -> consumer lanes, same wave)
  asm volatile("s_wait_dscnt 0" ::: "memory");
}

// A-fragment (16x32 f16, MxK) load from LDS staging (row stride 64 halfs).
// Layout per CDNA5 ISA 7.12.2: lanes 0-15 row M=lane, halfs j<8 -> K=koff+khalf+j,
// j>=8 -> K=koff+khalf+16+(j-8); lanes 16-31 same rows, khalf=8.
__device__ __forceinline__ v16h load_a_frag(const _Float16* rowBase, int lane,
                                            int koff) {
  const int row = lane & 15;
  const int khalf = (lane >> 4) << 3;
  const _Float16* p = rowBase + row * 64 + koff + khalf;
  v8h g0 = *(const v8h*)(p);
  v8h g1 = *(const v8h*)(p + 16);
  v16h a;
#pragma unroll
  for (int i = 0; i < 8; ++i) { a[i] = g0[i]; a[i + 8] = g1[i]; }
  return a;
}

// ---------------------------------------------------------------------------
// Kernel 0: zero output + degree counters
// ---------------------------------------------------------------------------
__global__ void zero_kernel(float* __restrict__ out, int n1,
                            float* __restrict__ cnt, int n2) {
  const int stride = gridDim.x * blockDim.x;
  for (int i = blockIdx.x * blockDim.x + threadIdx.x; i < n1; i += stride)
    out[i] = 0.0f;
  for (int i = blockIdx.x * blockDim.x + threadIdx.x; i < n2; i += stride)
    cnt[i] = 0.0f;
}

// ---------------------------------------------------------------------------
// Kernel 1: per-node atom MLP  Ai = silu(emb@W1+b1)@W2+b2   (10000 x 8)
// ---------------------------------------------------------------------------
__global__ __launch_bounds__(256) void node_kernel(
    const float* __restrict__ atom_emb, const float* __restrict__ w1,
    const float* __restrict__ b1, const float* __restrict__ w2,
    const float* __restrict__ b2, const int* __restrict__ A,
    float* __restrict__ Ai, int nNodes) {
  __shared__ float sw1[16 * 64];
  __shared__ float sb1[64];
  __shared__ float sw2[64 * 8];
  __shared__ float sb2[8];
  for (int i = threadIdx.x; i < 16 * 64; i += blockDim.x) sw1[i] = w1[i];
  for (int i = threadIdx.x; i < 64; i += blockDim.x) sb1[i] = b1[i];
  for (int i = threadIdx.x; i < 64 * 8; i += blockDim.x) sw2[i] = w2[i];
  if (threadIdx.x < 8) sb2[threadIdx.x] = b2[threadIdx.x];
  __syncthreads();

  const int n = blockIdx.x * blockDim.x + threadIdx.x;
  if (n >= nNodes) return;
  const int a = A[n];
  float emb[16];
#pragma unroll
  for (int k = 0; k < 16; ++k) emb[k] = atom_emb[a * 16 + k];
  float acc[8];
#pragma unroll
  for (int o = 0; o < 8; ++o) acc[o] = sb2[o];
#pragma unroll 4
  for (int j = 0; j < 64; ++j) {
    float h = sb1[j];
#pragma unroll
    for (int k = 0; k < 16; ++k) h += emb[k] * sw1[k * 64 + j];
    h = silu_f(h);
#pragma unroll
    for (int o = 0; o < 8; ++o) acc[o] += h * sw2[j * 8 + o];
  }
#pragma unroll
  for (int o = 0; o < 8; ++o) Ai[(size_t)n * 8 + o] = acc[o];
}

// ---------------------------------------------------------------------------
// Kernel 2: pre-swizzle all weight B-fragments into per-lane-contiguous f16.
// Fragment table (f = blockIdx.x, lane = threadIdx.x):
//   f 0..3   : W1 (K=32 padded from 10), ntile=f, kstep=0
//   f 4..11  : W2 (K=64), ks=(f-4)>>2, nt=(f-4)&3
//   f 12..13 : W3 (64 -> 16, 15 gates padded), ks=f-12
//   f 14..25 : T  (64 -> 96), nt=(f-14)>>1, ks=(f-14)&1
// B layout per ISA: lane holds col = lane&15, K = ks*32 + (lane>>4)*8 + j (j<8), +16+(j-8) (j>=8).
// ---------------------------------------------------------------------------
__global__ void prep_kernel(const float* __restrict__ fc_w1,
                            const float* __restrict__ fc_w2,
                            const float* __restrict__ fc_w3,
                            const float* __restrict__ tp_w,
                            _Float16* __restrict__ frag) {
  const int f = blockIdx.x;
  const int lane = threadIdx.x;
  const int col = lane & 15;
  const int hi = lane >> 4;
  _Float16 vals[16];
  if (f < 4) {
    const int n = f * 16 + col, kb = hi * 8;
#pragma unroll
    for (int j = 0; j < 16; ++j) {
      const int k = kb + j + ((j >> 3) << 3);
      vals[j] = (_Float16)((k < 10) ? fc_w1[k * 64 + n] : 0.0f);
    }
  } else if (f < 12) {
    const int idx = f - 4, ks = idx >> 2, nt = idx & 3;
    const int n = nt * 16 + col, kb = ks * 32 + hi * 8;
#pragma unroll
    for (int j = 0; j < 16; ++j) vals[j] = (_Float16)fc_w2[(kb + j + ((j >> 3) << 3)) * 64 + n];
  } else if (f < 14) {
    const int ks = f - 12, kb = ks * 32 + hi * 8;
#pragma unroll
    for (int j = 0; j < 16; ++j)
      vals[j] = (_Float16)((col < 15) ? fc_w3[(kb + j + ((j >> 3) << 3)) * 15 + col] : 0.0f);
  } else {
    const int idx = f - 14, nt = idx >> 1, ks = idx & 1;
    const int c = nt * 16 + col, pi = c >> 5, w = c & 31;
    const int p = (pi == 0) ? 0 : (pi == 1) ? 3 : 9;  // paths with l2==0
    const int kb = ks * 32 + hi * 8;
#pragma unroll
    for (int j = 0; j < 16; ++j) {
      const int k = kb + j + ((j >> 3) << 3), u = k >> 3, v = k & 7;
      vals[j] = (_Float16)tp_w[((p * 8 + u) * 8 + v) * 32 + w];
    }
  }
  v16h pack;
#pragma unroll
  for (int i = 0; i < 16; ++i) pack[i] = vals[i];
  ((v16h*)frag)[f * 32 + lane] = pack;
}

// ---------------------------------------------------------------------------
// Kernel 3: fused edge kernel. Each wave processes 32 edges (two 16-row WMMA
// M-tiles) per iteration of a grid-stride tile loop.
// ---------------------------------------------------------------------------
__global__ __launch_bounds__(256) void edge_kernel(
    const float* __restrict__ pos, const float* __restrict__ eshift,
    const float* __restrict__ cell, const float* __restrict__ fcb1,
    const float* __restrict__ fcb2, const float* __restrict__ fcb3,
    const int* __restrict__ batch, const int* __restrict__ esrc,
    const int* __restrict__ edst, const float* __restrict__ Ai,
    const _Float16* __restrict__ frag, float* __restrict__ out,
    float* __restrict__ cnt, int nTiles) {
  __shared__ __attribute__((aligned(16))) _Float16 stageH[WAVES_PER_BLOCK][32 * 64];
  __shared__ __attribute__((aligned(16))) _Float16 aiH[WAVES_PER_BLOCK][32 * 16];
  __shared__ __attribute__((aligned(16))) float    coefF[WAVES_PER_BLOCK][32 * 12];
  __shared__ int dstI[WAVES_PER_BLOCK][32];

  const int lane = threadIdx.x & 31;
  const int wv = threadIdx.x >> 5;
  const int colB = lane & 15;
  const int hi = lane >> 4;

  // Resident B fragments (loaded once, reused over the tile loop).
  v16h bw1[4], bw2[8], bw3[2], bT[12];
#pragma unroll
  for (int t = 0; t < 4; ++t)
    bw1[t] = *(const v16h*)(frag + ((t) * 32 + lane) * 16);
#pragma unroll
  for (int i = 0; i < 8; ++i)
    bw2[i] = *(const v16h*)(frag + ((4 + i) * 32 + lane) * 16);
#pragma unroll
  for (int i = 0; i < 2; ++i)
    bw3[i] = *(const v16h*)(frag + ((12 + i) * 32 + lane) * 16);
#pragma unroll
  for (int i = 0; i < 12; ++i)
    bT[i] = *(const v16h*)(frag + ((14 + i) * 32 + lane) * 16);

  float b1v[4], b2v[4];
#pragma unroll
  for (int t = 0; t < 4; ++t) {
    b1v[t] = fcb1[t * 16 + colB];
    b2v[t] = fcb2[t * 16 + colB];
  }
  const float b3v = (colB < 15) ? fcb3[colB] : 0.0f;

  const int waveGlobal = blockIdx.x * WAVES_PER_BLOCK + wv;
  const int waveCount = gridDim.x * WAVES_PER_BLOCK;
  const float step = 5.0f / 11.0f;
  const float rstep = 11.0f / 5.0f;

  for (int tile = waveGlobal; tile < nTiles; tile += waveCount) {
    const int e = tile * 32 + lane;
    if (tile + waveCount < nTiles) {  // speculative prefetch of next tile idx
      __builtin_prefetch(esrc + (tile + waveCount) * 32 + lane, 0, 1);
      __builtin_prefetch(edst + (tile + waveCount) * 32 + lane, 0, 1);
    }

    // ---------------- scalar per-edge phase (one edge per lane) ------------
    const int s = esrc[e];
    const int d = edst[e];
    const float px = pos[s * 3 + 0], py = pos[s * 3 + 1], pz = pos[s * 3 + 2];
    const float qx = pos[d * 3 + 0], qy = pos[d * 3 + 1], qz = pos[d * 3 + 2];
    const int bg = batch[s];
    const float sx = eshift[(size_t)e * 3 + 0];
    const float sy = eshift[(size_t)e * 3 + 1];
    const float sz = eshift[(size_t)e * 3 + 2];
    const float* C = cell + bg * 9;
    const float shx = sx * C[0] + sy * C[3] + sz * C[6];
    const float shy = sx * C[1] + sy * C[4] + sz * C[7];
    const float shz = sx * C[2] + sy * C[5] + sz * C[8];
    const float vx = qx - px + shx, vy = qy - py + shy, vz = qz - pz + shz;
    const float r = sqrtf(vx * vx + vy * vy + vz * vz);
    const float inv = 1.0f / fmaxf(r, 1e-8f);
    const float nx = vx * inv, ny = vy * inv, nz = vz * inv;
    // spherical harmonics
    const float Y1x = 1.7320508f * nx, Y1y = 1.7320508f * ny,
                Y1z = 1.7320508f * nz;
    const float a15 = 3.8729833f;
    const float Y20 = a15 * nx * ny;
    const float Y21 = a15 * ny * nz;
    const float Y22 = 1.1180340f * (3.0f * nz * nz - 1.0f);
    const float Y23 = a15 * nx * nz;
    const float Y24 = 1.9364917f * (nx * nx - ny * ny);

    // radial basis -> A-fragment staging rows (K padded 10 -> 32)
#pragma unroll
    for (int i = 0; i < 10; ++i) {
      const float dd = (r - (float)(i + 1) * step) * rstep;
      stageH[wv][lane * 64 + i] = (_Float16)(__expf(-dd * dd) * 2.8234622f);
    }
#pragma unroll
    for (int i = 10; i < 32; ++i) stageH[wv][lane * 64 + i] = (_Float16)0.0f;

    // gather Ai for src/dst, stage as f16
    v4f as0 = *(const v4f*)(Ai + (size_t)s * 8);
    v4f as1 = *(const v4f*)(Ai + (size_t)s * 8 + 4);
    v4f ad0 = *(const v4f*)(Ai + (size_t)d * 8);
    v4f ad1 = *(const v4f*)(Ai + (size_t)d * 8 + 4);
#pragma unroll
    for (int i = 0; i < 4; ++i) {
      aiH[wv][lane * 16 + i] = (_Float16)as0[i];
      aiH[wv][lane * 16 + 4 + i] = (_Float16)as1[i];
      aiH[wv][lane * 16 + 8 + i] = (_Float16)ad0[i];
      aiH[wv][lane * 16 + 12 + i] = (_Float16)ad1[i];
    }
    dstI[wv][lane] = d;
    atomicAdd(&cnt[d], 1.0f);
    lds_fence_wave();

    // ---------------- FC layer 1: (32e x 32) @ (32 x 64) -------------------
    v16h a1[2];
#pragma unroll
    for (int mt = 0; mt < 2; ++mt)
      a1[mt] = load_a_frag(&stageH[wv][mt * 16 * 64], lane, 0);
    v8f acc1[2][4];
#pragma unroll
    for (int mt = 0; mt < 2; ++mt)
#pragma unroll
      for (int nt = 0; nt < 4; ++nt) {
        v8f c;
#pragma unroll
        for (int i = 0; i < 8; ++i) c[i] = b1v[nt];
        acc1[mt][nt] = wmma_f16(a1[mt], bw1[nt], c);
      }
    // silu + restage as f16 (overwrites rb region; A-frags already read)
#pragma unroll
    for (int mt = 0; mt < 2; ++mt)
#pragma unroll
      for (int nt = 0; nt < 4; ++nt) {
        const int col = nt * 16 + colB;
#pragma unroll
        for (int rr = 0; rr < 8; ++rr)
          stageH[wv][(mt * 16 + hi * 8 + rr) * 64 + col] =
              (_Float16)silu_f(acc1[mt][nt][rr]);
      }
    lds_fence_wave();

    // ---------------- FC layer 2: (32e x 64) @ (64 x 64) -------------------
    v16h a2[2][2];
#pragma unroll
    for (int mt = 0; mt < 2; ++mt)
#pragma unroll
      for (int ks = 0; ks < 2; ++ks)
        a2[mt][ks] = load_a_frag(&stageH[wv][mt * 16 * 64], lane, ks * 32);
    v8f acc2[2][4];
#pragma unroll
    for (int mt = 0; mt < 2; ++mt)
#pragma unroll
      for (int nt = 0; nt < 4; ++nt) {
        v8f c;
#pragma unroll
        for (int i = 0; i < 8; ++i) c[i] = b2v[nt];
        c = wmma_f16(a2[mt][0], bw2[0 + nt], c);
        c = wmma_f16(a2[mt][1], bw2[4 + nt], c);
        acc2[mt][nt] = c;
      }
#pragma unroll
    for (int mt = 0; mt < 2; ++mt)
#pragma unroll
      for (int nt = 0; nt < 4; ++nt) {
        const int col = nt * 16 + colB;
#pragma unroll
        for (int rr = 0; rr < 8; ++rr)
          stageH[wv][(mt * 16 + hi * 8 + rr) * 64 + col] =
              (_Float16)silu_f(acc2[mt][nt][rr]);
      }
    lds_fence_wave();

    // ---------------- FC layer 3 (gates): (32e x 64) @ (64 x 16) -----------
    v16h a3[2][2];
#pragma unroll
    for (int mt = 0; mt < 2; ++mt)
#pragma unroll
      for (int ks = 0; ks < 2; ++ks)
        a3[mt][ks] = load_a_frag(&stageH[wv][mt * 16 * 64], lane, ks * 32);
    v8f gacc[2];
#pragma unroll
    for (int mt = 0; mt < 2; ++mt) {
      v8f c;
#pragma unroll
      for (int i = 0; i < 8; ++i) c[i] = b3v;
      c = wmma_f16(a3[mt][0], bw3[0], c);
      c = wmma_f16(a3[mt][1], bw3[1], c);
      gacc[mt] = c;
    }
    // stash the three relevant gates (paths 0,3,9) per edge row
    if (colB == 0 || colB == 3 || colB == 9) {
      const int c3 = (colB == 0) ? 0 : (colB == 3) ? 1 : 2;
#pragma unroll
      for (int mt = 0; mt < 2; ++mt)
#pragma unroll
        for (int rr = 0; rr < 8; ++rr)
          coefF[wv][(mt * 16 + hi * 8 + rr) * 12 + 9 + c3] = gacc[mt][rr];
    }
    lds_fence_wave();

    // build gate*Y coefficients for own edge (row == lane)
    {
      const float g0 = coefF[wv][lane * 12 + 9];
      const float g1 = coefF[wv][lane * 12 + 10];
      const float g2 = coefF[wv][lane * 12 + 11];
      coefF[wv][lane * 12 + 0] = g0;
      coefF[wv][lane * 12 + 1] = g1 * Y1x;
      coefF[wv][lane * 12 + 2] = g1 * Y1y;
      coefF[wv][lane * 12 + 3] = g1 * Y1z;
      coefF[wv][lane * 12 + 4] = g2 * Y20;
      coefF[wv][lane * 12 + 5] = g2 * Y21;
      coefF[wv][lane * 12 + 6] = g2 * Y22;
      coefF[wv][lane * 12 + 7] = g2 * Y23;
      coefF[wv][lane * 12 + 8] = g2 * Y24;
    }
    lds_fence_wave();

    // ---------------- tensor product + scatter, one M-tile at a time -------
    // A-fragment element j   <-> K = ks*32 + hi*8 + j      -> u = ks*4+hi,   v=j
    // A-fragment element 8+j <-> K = ks*32 + hi*8 + 16 + j -> u = ks*4+hi+2, v=j
    // so frag = concat( s[u0]*dvec , s[u1]*dvec ) with only scalar s loads
    // (no dynamic vector extracts -> pure v_pk_mul_f16).
#pragma unroll
    for (int mt = 0; mt < 2; ++mt) {
      const int m = mt * 16 + colB;
      const v8h dvec = *(const v8h*)(&aiH[wv][m * 16 + 8]);
      v16h aT[2];
#pragma unroll
      for (int ks = 0; ks < 2; ++ks) {
        const _Float16 s0 = aiH[wv][m * 16 + ks * 4 + hi];
        const _Float16 s1 = aiH[wv][m * 16 + ks * 4 + hi + 2];
        v16h a;
#pragma unroll
        for (int j = 0; j < 8; ++j) {
          a[j] = s0 * dvec[j];
          a[8 + j] = s1 * dvec[j];
        }
        aT[ks] = a;
      }
      v8f accT[6];
#pragma unroll
      for (int t = 0; t < 6; ++t) {
        v8f c;
#pragma unroll
        for (int i = 0; i < 8; ++i) c[i] = 0.0f;
        c = wmma_f16(aT[0], bT[t * 2 + 0], c);
        c = wmma_f16(aT[1], bT[t * 2 + 1], c);
        accT[t] = c;
      }

      // gate*Y scaling + atomic scatter to out[dst]
#pragma unroll
      for (int rr = 0; rr < 8; ++rr) {
        const int grow = mt * 16 + hi * 8 + rr;
        const int dn = dstI[wv][grow];
        float* op = out + (size_t)dn * NFEAT;
        const float* cf = &coefF[wv][grow * 12];
        const float c0 = cf[0];
        atomicAdd(op + colB, c0 * accT[0][rr]);
        atomicAdd(op + 16 + colB, c0 * accT[1][rr]);
#pragma unroll
        for (int k = 0; k < 3; ++k) {
          const float ck = cf[1 + k];
          atomicAdd(op + 32 + colB * 3 + k, ck * accT[2][rr]);
          atomicAdd(op + 32 + (16 + colB) * 3 + k, ck * accT[3][rr]);
        }
#pragma unroll
        for (int k = 0; k < 5; ++k) {
          const float ck = cf[4 + k];
          atomicAdd(op + 128 + colB * 5 + k, ck * accT[4][rr]);
          atomicAdd(op + 128 + (16 + colB) * 5 + k, ck * accT[5][rr]);
        }
      }
    }
  }
}

// ---------------------------------------------------------------------------
// Kernel 4: divide by in-degree (min 1)
// ---------------------------------------------------------------------------
__global__ void norm_kernel(float* __restrict__ out,
                            const float* __restrict__ cnt, int total) {
  const int i = blockIdx.x * blockDim.x + threadIdx.x;
  if (i >= total) return;
  const int n = i / NFEAT;
  out[i] = out[i] / fmaxf(cnt[n], 1.0f);
}

// ---------------------------------------------------------------------------
// Host-side launcher
// ---------------------------------------------------------------------------
extern "C" void kernel_launch(void* const* d_in, const int* in_sizes, int n_in,
                              void* d_out, int out_size, void* d_ws,
                              size_t ws_size, hipStream_t stream) {
  const float* pos = (const float*)d_in[0];
  const float* eshift = (const float*)d_in[1];
  const float* cell = (const float*)d_in[2];
  const float* atom_emb = (const float*)d_in[3];
  const float* mlp_w1 = (const float*)d_in[4];
  const float* mlp_b1 = (const float*)d_in[5];
  const float* mlp_w2 = (const float*)d_in[6];
  const float* mlp_b2 = (const float*)d_in[7];
  const float* fc_w1 = (const float*)d_in[8];
  const float* fc_b1 = (const float*)d_in[9];
  const float* fc_w2 = (const float*)d_in[10];
  const float* fc_b2 = (const float*)d_in[11];
  const float* fc_w3 = (const float*)d_in[12];
  const float* fc_b3 = (const float*)d_in[13];
  const float* tp_w = (const float*)d_in[14];
  const int* A = (const int*)d_in[15];
  const int* batch = (const int*)d_in[16];
  const int* esrc = (const int*)d_in[17];
  const int* edst = (const int*)d_in[18];
  float* out = (float*)d_out;

  const int nNodes = in_sizes[15];
  const int nEdges = in_sizes[17];
  const int nTiles = nEdges / 32;

  // workspace layout (floats): [0) Ai 8*nNodes | cnt nNodes | f16 fragments
  float* wsF = (float*)d_ws;
  float* Ai = wsF;
  float* cnt = wsF + (size_t)nNodes * 8;
  _Float16* frag = (_Float16*)(wsF + (size_t)nNodes * 8 + nNodes);

  zero_kernel<<<2048, 256, 0, stream>>>(out, out_size, cnt, nNodes);
  node_kernel<<<(nNodes + 255) / 256, 256, 0, stream>>>(
      atom_emb, mlp_w1, mlp_b1, mlp_w2, mlp_b2, A, Ai, nNodes);
  prep_kernel<<<26, 32, 0, stream>>>(fc_w1, fc_w2, fc_w3, tp_w, frag);
  edge_kernel<<<160, 256, 0, stream>>>(pos, eshift, cell, fc_b1, fc_b2, fc_b3,
                                       batch, esrc, edst, Ai, frag, out, cnt,
                                       nTiles);
  norm_kernel<<<(out_size + 255) / 256, 256, 0, stream>>>(out, cnt, out_size);
}
